// HeavyCausalSelfAttention_7378753814618
// MI455X (gfx1250) — matrix operand order repaired and checked by reading, not compiled, not run on hardware
//
#include <hip/hip_runtime.h>
#include <cstdint>
#include <cstddef>

// ---------------------------------------------------------------------------
// Problem constants (match the JAX reference)
// ---------------------------------------------------------------------------
constexpr int Bc  = 2;
constexpr int Tc  = 1024;
constexpr int Cc  = 1024;
constexpr int NHc = 16;
constexpr int NGc = 2;
constexpr int HSc = 64;   // Cc / NHc

typedef __bf16 bf16;
typedef bf16  v16bf __attribute__((ext_vector_type(16)));
typedef float v8f   __attribute__((ext_vector_type(8)));

#define DEVINL __device__ __forceinline__

// ---------------------------------------------------------------------------
// gfx1250 async Global->LDS copy (DMA path, ASYNCcnt-tracked) with fallback
// ---------------------------------------------------------------------------
#ifdef __has_builtin
#if __has_builtin(__builtin_amdgcn_global_load_async_to_lds_b128)
#define HAVE_ASYNC_LDS 1
#endif
#endif
#ifndef HAVE_ASYNC_LDS
#define HAVE_ASYNC_LDS 0
#endif

// Builtin parameter types (from hipcc diagnostic): AS(1)/AS(3) pointers to a
// 16-byte GCC vector of int.
typedef int gcc_v4i __attribute__((vector_size(16)));
typedef __attribute__((address_space(1))) gcc_v4i* g_v4i_ptr;
typedef __attribute__((address_space(3))) gcc_v4i* l_v4i_ptr;

DEVINL void cp_async_b128(bf16* lds_dst, const bf16* gsrc) {
#if HAVE_ASYNC_LDS
  __builtin_amdgcn_global_load_async_to_lds_b128(
      (g_v4i_ptr)gsrc, (l_v4i_ptr)lds_dst, /*offset=*/0, /*cpol=*/0);
#else
  *(uint4*)lds_dst = *(const uint4*)gsrc;
#endif
}

DEVINL void wait_async0() {
#if HAVE_ASYNC_LDS
#if __has_builtin(__builtin_amdgcn_s_wait_asynccnt)
  __builtin_amdgcn_s_wait_asynccnt(0);
#else
  asm volatile("s_wait_asynccnt 0x0" ::: "memory");
#endif
#endif
}

DEVINL bf16 f2bf(float f) {
  unsigned u = __float_as_uint(f);
  u += 0x7fffu + ((u >> 16) & 1u);          // round-to-nearest-even
  unsigned short s = (unsigned short)(u >> 16);
  return __builtin_bit_cast(bf16, s);
}

DEVINL v8f wmma_bf16(v16bf a, v16bf b, v8f c) {
  // D = A(16x32 bf16) * B(32x16 bf16) + C(16x16 f32)
  return __builtin_amdgcn_wmma_f32_16x16x32_bf16(
      /*neg_a=*/false, a, /*neg_b=*/false, b,
      /*c_mod=*/(short)0, c, /*reuse_a=*/false, /*reuse_b=*/false);
}

// A-fragment (16x32, 16-bit): lane m = lane&15 holds row m.
//   v0..v7 : K = kbase + khalf*16 .. +15 (32 bytes, contiguous)
DEVINL v16bf a_frag(const bf16* p, int stride, int lane, int kbase) {
  const int m = lane & 15, kh = lane >> 4;
  union { v16bf v; uint4 q[2]; } u;
  const bf16* s = p + (size_t)m * stride + kbase + kh * 16;
  u.q[0] = *(const uint4*)(s);
  u.q[1] = *(const uint4*)(s + 8);
  return u.v;
}

// B-fragment (32x16, 16-bit) from column-contiguous storage (N-major):
// lane n = lane&15 holds column n, K = khalf*16 .. +15 (32 contiguous bytes).
DEVINL v16bf b_frag(const bf16* p, int stride, int lane) {
  const int nn = lane & 15, kh = lane >> 4;
  union { v16bf v; uint4 q[2]; } u;
  const bf16* s = p + (size_t)nn * stride + kh * 16;
  u.q[0] = *(const uint4*)(s);
  u.q[1] = *(const uint4*)(s + 8);
  return u.v;
}

// ---------------------------------------------------------------------------
// Kernel 1: weight transpose + convert  Wt[n][k] = bf16(W[k][n])
// ---------------------------------------------------------------------------
__global__ __launch_bounds__(256)
void wt_cvt_kernel(const float* __restrict__ W, bf16* __restrict__ Wt,
                   int K, int N) {
  __shared__ float tile[16][17];
  const int n0 = blockIdx.x * 16, k0 = blockIdx.y * 16;
  const int tx = threadIdx.x & 15, ty = threadIdx.x >> 4;
  tile[ty][tx] = W[(size_t)(k0 + ty) * N + n0 + tx];
  __syncthreads();
  Wt[(size_t)(n0 + ty) * K + k0 + tx] = f2bf(tile[tx][ty]);
}

// ---------------------------------------------------------------------------
// Kernel 2: activation convert f32 -> bf16 (4 elements / thread)
// ---------------------------------------------------------------------------
__global__ __launch_bounds__(256)
void x_cvt_kernel(const float* __restrict__ x, bf16* __restrict__ y) {
  const size_t i = ((size_t)blockIdx.x * 256 + threadIdx.x) * 4;
  const float4 v = *(const float4*)(x + i);
  union { bf16 b[4]; uint2 u; } o;
  o.b[0] = f2bf(v.x); o.b[1] = f2bf(v.y);
  o.b[2] = f2bf(v.z); o.b[3] = f2bf(v.w);
  *(uint2*)(y + i) = o.u;
}

// ---------------------------------------------------------------------------
// Kernel 3: C(f32, MxN) = A(bf16, MxK) * Bt(bf16, NxK)^T
// Block tile 128x128, 8 waves (4 in M, 2 in N); each wave a 32x64 WMMA tile
// (8 v_wmma per 32-wide K chunk). Global->LDS staging uses the gfx1250
// async-DMA path when available.
// ---------------------------------------------------------------------------
__global__ __launch_bounds__(256)
void gemm_bf16nt_kernel(const bf16* __restrict__ A, const bf16* __restrict__ Bt,
                        float* __restrict__ Cout, int M, int N, int K) {
  __shared__ bf16 lA[128][40];   // pitch 80B (multiple of 16B, conflict-friendly)
  __shared__ bf16 lB[128][40];
  const int tid  = threadIdx.x;
  const int lane = tid & 31, wave = tid >> 5;
  const int wm = wave & 3, wn = wave >> 2;          // 4 x 2 wave grid
  const int m0 = blockIdx.x * 128, n0 = blockIdx.y * 128;
  v8f acc[2][4] = {};

  // 128 rows x 32 cols per matrix; 2 threads/row, 32B per thread per matrix
  const int r = tid >> 1, c = (tid & 1) * 16;
  const bf16* gA = A  + (size_t)(m0 + r) * K + c;
  const bf16* gB = Bt + (size_t)(n0 + r) * K + c;
  bf16* sA = &lA[r][c];
  bf16* sB = &lB[r][c];

  for (int k0 = 0; k0 < K; k0 += 32) {
    cp_async_b128(sA,     gA + k0);
    cp_async_b128(sA + 8, gA + k0 + 8);
    cp_async_b128(sB,     gB + k0);
    cp_async_b128(sB + 8, gB + k0 + 8);
    if (k0 + 32 < K) {
      __builtin_prefetch(gA + k0 + 32, 0, 1);
      __builtin_prefetch(gB + k0 + 32, 0, 1);
    }
    wait_async0();
    __syncthreads();
    const v16bf a0 = a_frag(&lA[wm * 32][0],      40, lane, 0);
    const v16bf a1 = a_frag(&lA[wm * 32 + 16][0], 40, lane, 0);
#pragma unroll
    for (int j = 0; j < 4; ++j) {
      const v16bf bf = b_frag(&lB[wn * 64 + j * 16][0], 40, lane);
      acc[0][j] = wmma_bf16(a0, bf, acc[0][j]);
      acc[1][j] = wmma_bf16(a1, bf, acc[1][j]);
    }
    __syncthreads();
  }

  // C layout: lane n = lane&15 holds column, VGPR r -> row r + 8*(lane>>4)
  const int nl = lane & 15, hw = lane >> 4;
#pragma unroll
  for (int i = 0; i < 2; ++i) {
#pragma unroll
    for (int j = 0; j < 4; ++j) {
      const int col  = n0 + wn * 64 + j * 16 + nl;
      const int row0 = m0 + wm * 32 + i * 16 + 8 * hw;
#pragma unroll
      for (int rr = 0; rr < 8; ++rr)
        Cout[(size_t)(row0 + rr) * N + col] = acc[i][j][rr];
    }
  }
}

// ---------------------------------------------------------------------------
// Kernel 4: fused RMSNorm + RoPE; emits head-major bf16 Q/K and transposed V
// ---------------------------------------------------------------------------
DEVINL float block_sum256(float v) {
  __shared__ float sbuf[9];
#pragma unroll
  for (int m = 16; m >= 1; m >>= 1) v += __shfl_xor(v, m, 32);
  __syncthreads();                      // guard sbuf reuse across calls
  if ((threadIdx.x & 31) == 0) sbuf[threadIdx.x >> 5] = v;
  __syncthreads();
  if (threadIdx.x == 0) {
    float t = 0.f;
    for (int i = 0; i < 8; ++i) t += sbuf[i];
    sbuf[8] = t;
  }
  __syncthreads();
  return sbuf[8];
}

DEVINL void rope_store4(const float4& x, float rms, const float4& w,
                        int t, int d0, bf16* base, size_t hid0) {
  const float xn[4] = { x.x * rms * w.x, x.y * rms * w.y,
                        x.z * rms * w.z, x.w * rms * w.w };
  constexpr float kLn1e4_64 = 9.210340371976184f / 64.f;
#pragma unroll
  for (int pp = 0; pp < 2; ++pp) {
    const int d  = d0 + 2 * pp;
    const int h  = d >> 6, hs = d & 63;
    const int j0 = hs & 31, j1 = (hs + 1) & 31;
    const float a0 = (float)t * __expf(-(4.f * j0 + 1.f) * kLn1e4_64);
    const float a1 = (float)t * __expf(-(4.f * j1 + 1.f) * kLn1e4_64);
    const float e = xn[2 * pp], o = xn[2 * pp + 1];
    const float y0 = e * __cosf(a0) - o * __sinf(a0);
    const float y1 = o * __cosf(a1) + e * __sinf(a1);
    const size_t idx = ((hid0 + h) * (size_t)Tc + t) * HSc + hs;
    base[idx]     = f2bf(y0);
    base[idx + 1] = f2bf(y1);
  }
}

__global__ __launch_bounds__(256)
void normrope_kernel(const float* __restrict__ qf, const float* __restrict__ kvf,
                     const float* __restrict__ lnw,
                     bf16* __restrict__ qb, bf16* __restrict__ kb,
                     bf16* __restrict__ vt) {
  const int bt = blockIdx.x;
  const int b = bt / Tc, t = bt % Tc;
  const float* qrow  = qf  + (size_t)bt * (NGc * Cc);
  const float* kvrow = kvf + (size_t)bt * (2 * Cc);
  const int d0 = threadIdx.x * 4;

  const float4 q0 = *(const float4*)(qrow + d0);
  const float4 q1 = *(const float4*)(qrow + Cc + d0);
  const float4 kk = *(const float4*)(kvrow + d0);
  const float4 vv = *(const float4*)(kvrow + Cc + d0);

  const float r0 = rsqrtf(block_sum256(q0.x*q0.x + q0.y*q0.y + q0.z*q0.z + q0.w*q0.w) * (1.f / Cc) + 1e-6f);
  const float r1 = rsqrtf(block_sum256(q1.x*q1.x + q1.y*q1.y + q1.z*q1.z + q1.w*q1.w) * (1.f / Cc) + 1e-6f);
  const float rk = rsqrtf(block_sum256(kk.x*kk.x + kk.y*kk.y + kk.z*kk.z + kk.w*kk.w) * (1.f / Cc) + 1e-6f);

  const float4 w4 = *(const float4*)(lnw + d0);
  rope_store4(q0, r0, w4, t, d0, qb, (size_t)(b * NGc + 0) * NHc);
  rope_store4(q1, r1, w4, t, d0, qb, (size_t)(b * NGc + 1) * NHc);
  rope_store4(kk, rk, w4, t, d0, kb, (size_t)b * NHc);

  // V transposed: vt[(b,h)][hs][t] so att@V B-fragments are contiguous loads
  const float vvv[4] = { vv.x, vv.y, vv.z, vv.w };
#pragma unroll
  for (int e = 0; e < 4; ++e) {
    const int d = d0 + e, h = d >> 6, hs = d & 63;
    vt[(((size_t)b * NHc + h) * HSc + hs) * Tc + t] = f2bf(vvv[e]);
  }
}

// ---------------------------------------------------------------------------
// Kernel 5: flash-style causal attention, one wave per 16-query tile.
// ---------------------------------------------------------------------------
__global__ __launch_bounds__(128)
void attn_kernel(const bf16* __restrict__ Q, const bf16* __restrict__ K,
                 const bf16* __restrict__ V, bf16* __restrict__ Y) {
  __shared__ bf16 ldsP[4][16][40];      // per-wave P staging (C-layout -> A-layout)
  const int lane = threadIdx.x & 31, wave = threadIdx.x >> 5;
  const int hid = blockIdx.x;           // (b*NG + g)*NH + h, 0..63
  const int b = hid >> 5, g = (hid >> 4) & 1, h = hid & 15;
  const int t0 = (blockIdx.y * 4 + wave) * 16;
  const bf16* qp = Q + ((size_t)hid * Tc + t0) * HSc;
  const bf16* kp = K + (size_t)(b * NHc + h) * Tc * HSc;
  const bf16* vp = V + (size_t)(b * NHc + h) * HSc * Tc;
  const int n = lane & 15, hw = lane >> 4;

  const v16bf aQ0 = a_frag(qp, HSc, lane, 0);
  const v16bf aQ1 = a_frag(qp, HSc, lane, 32);

  v8f o[4] = {};
  float mi[8], li[8];
#pragma unroll
  for (int r = 0; r < 8; ++r) { mi[r] = -3e38f; li[r] = 0.f; }

  const int nch = (t0 + 47) >> 5;       // 32-key chunks covering keys 0..t0+15
  for (int cc = 0; cc < nch; ++cc) {
    const int k0 = cc * 32;

    // scores S = Q * K^T (two 16x16 tiles per chunk)
    v8f sa = {}, sb = {};
    {
      const v16bf b00 = b_frag(kp + (size_t)k0 * HSc,        HSc, lane);
      const v16bf b01 = b_frag(kp + (size_t)k0 * HSc + 32,   HSc, lane);
      sa = wmma_bf16(aQ0, b00, sa);
      sa = wmma_bf16(aQ1, b01, sa);
      const v16bf b10 = b_frag(kp + (size_t)(k0 + 16) * HSc,      HSc, lane);
      const v16bf b11 = b_frag(kp + (size_t)(k0 + 16) * HSc + 32, HSc, lane);
      sb = wmma_bf16(aQ0, b10, sb);
      sb = wmma_bf16(aQ1, b11, sb);
    }

    // scale + causal mask + online softmax (row r lives at VGPR r, row t0+r+8*hw)
    float pa[8], pb[8];
#pragma unroll
    for (int r = 0; r < 8; ++r) {
      const int qrow = t0 + r + 8 * hw;
      float va = (k0 + n      <= qrow) ? sa[r] * 0.125f : -3e38f;
      float vb = (k0 + 16 + n <= qrow) ? sb[r] * 0.125f : -3e38f;
      float mx = fmaxf(va, vb);
      mx = fmaxf(mx, __shfl_xor(mx, 1, 32));
      mx = fmaxf(mx, __shfl_xor(mx, 2, 32));
      mx = fmaxf(mx, __shfl_xor(mx, 4, 32));
      mx = fmaxf(mx, __shfl_xor(mx, 8, 32));
      const float nm = fmaxf(mi[r], mx);
      const float al = __expf(mi[r] - nm);
      pa[r] = __expf(va - nm);
      pb[r] = __expf(vb - nm);
      float rs = pa[r] + pb[r];
      rs += __shfl_xor(rs, 1, 32);
      rs += __shfl_xor(rs, 2, 32);
      rs += __shfl_xor(rs, 4, 32);
      rs += __shfl_xor(rs, 8, 32);
      li[r] = li[r] * al + rs;
      mi[r] = nm;
#pragma unroll
      for (int j = 0; j < 4; ++j) o[j][r] *= al;
    }

    // C-layout P -> LDS -> A-layout fragment (16x32 bf16)
#pragma unroll
    for (int r = 0; r < 8; ++r) {
      ldsP[wave][r + 8 * hw][n]      = f2bf(pa[r]);
      ldsP[wave][r + 8 * hw][16 + n] = f2bf(pb[r]);
    }
    const v16bf aP  = a_frag(&ldsP[wave][0][0], 40, lane, 0);
    const v16bf bv0 = b_frag(vp + k0,                     Tc, lane);
    const v16bf bv1 = b_frag(vp + (size_t)16 * Tc + k0,   Tc, lane);
    const v16bf bv2 = b_frag(vp + (size_t)32 * Tc + k0,   Tc, lane);
    const v16bf bv3 = b_frag(vp + (size_t)48 * Tc + k0,   Tc, lane);
    o[0] = wmma_bf16(aP, bv0, o[0]);
    o[1] = wmma_bf16(aP, bv1, o[1]);
    o[2] = wmma_bf16(aP, bv2, o[2]);
    o[3] = wmma_bf16(aP, bv3, o[3]);
  }

  // epilogue: y[b][g][t][h*64 + hs] (bf16, feeds the final GEMM as A matrix)
  bf16* yp = Y + ((size_t)(b * NGc + g) * Tc) * Cc;
#pragma unroll
  for (int r = 0; r < 8; ++r) {
    const float inv = 1.f / li[r];
    const size_t row = (size_t)(t0 + r + 8 * hw) * Cc + h * HSc + n;
#pragma unroll
    for (int j = 0; j < 4; ++j) yp[row + (size_t)j * 16] = f2bf(o[j][r] * inv);
  }
}

// ---------------------------------------------------------------------------
// Launcher
// ---------------------------------------------------------------------------
extern "C" void kernel_launch(void* const* d_in, const int* in_sizes, int n_in,
                              void* d_out, int out_size, void* d_ws, size_t ws_size,
                              hipStream_t stream) {
  (void)in_sizes; (void)n_in; (void)out_size; (void)ws_size;
  const float* query_x = (const float*)d_in[0];
  const float* kv_x    = (const float*)d_in[1];
  // d_in[2]: causal mask (bool) -- causality is computed analytically, unused
  const float* Wq    = (const float*)d_in[3];
  const float* Wkv   = (const float*)d_in[4];
  const float* Wproj = (const float*)d_in[5];
  const float* lnw   = (const float*)d_in[6];
  float* out = (float*)d_out;

  char* ws = (char*)d_ws;
  size_t off = 0;
  auto carve = [&](size_t bytes) -> void* {
    void* p = ws + off;
    off += (bytes + 255) & ~(size_t)255;
    return p;
  };
  bf16*  WqT  = (bf16*)carve((size_t)NGc * Cc * Cc * 2);
  bf16*  WkvT = (bf16*)carve((size_t)2 * Cc * Cc * 2);
  bf16*  WpT  = (bf16*)carve((size_t)Cc * Cc * 2);
  bf16*  xqB  = (bf16*)carve((size_t)Bc * Tc * Cc * 2);
  bf16*  xkvB = (bf16*)carve((size_t)Bc * Tc * Cc * 2);
  float* qF   = (float*)carve((size_t)Bc * Tc * NGc * Cc * 4);
  float* kvF  = (float*)carve((size_t)Bc * Tc * 2 * Cc * 4);
  bf16*  qB   = (bf16*)carve((size_t)Bc * NGc * NHc * Tc * HSc * 2);
  bf16*  kB   = (bf16*)carve((size_t)Bc * NHc * Tc * HSc * 2);
  bf16*  vT   = (bf16*)carve((size_t)Bc * NHc * HSc * Tc * 2);
  bf16*  yB   = (bf16*)carve((size_t)Bc * NGc * Tc * Cc * 2);

  // 1) weights: transpose + bf16
  wt_cvt_kernel<<<dim3((NGc * Cc) / 16, Cc / 16), 256, 0, stream>>>(Wq,    WqT,  Cc, NGc * Cc);
  wt_cvt_kernel<<<dim3((2 * Cc)   / 16, Cc / 16), 256, 0, stream>>>(Wkv,   WkvT, Cc, 2 * Cc);
  wt_cvt_kernel<<<dim3(Cc / 16,        Cc / 16), 256, 0, stream>>>(Wproj, WpT,  Cc, Cc);

  // 2) activations -> bf16
  x_cvt_kernel<<<(Bc * Tc * Cc) / 1024, 256, 0, stream>>>(query_x, xqB);
  x_cvt_kernel<<<(Bc * Tc * Cc) / 1024, 256, 0, stream>>>(kv_x,    xkvB);

  // 3) q = x*Wq, kv = x*Wkv  (f32 outputs for the norm pass)
  gemm_bf16nt_kernel<<<dim3((Bc * Tc) / 128, (NGc * Cc) / 128), 256, 0, stream>>>(
      xqB, WqT, qF, Bc * Tc, NGc * Cc, Cc);
  gemm_bf16nt_kernel<<<dim3((Bc * Tc) / 128, (2 * Cc) / 128), 256, 0, stream>>>(
      xkvB, WkvT, kvF, Bc * Tc, 2 * Cc, Cc);

  // 4) RMSNorm + RoPE -> head-major bf16 Q/K, transposed bf16 V
  normrope_kernel<<<Bc * Tc, 256, 0, stream>>>(qF, kvF, lnw, qB, kB, vT);

  // 5) causal flash attention -> y (bf16)
  attn_kernel<<<dim3(Bc * NGc * NHc, Tc / 64), 128, 0, stream>>>(qB, kB, vT, yB);

  // 6) out = y * Wproj (f32, straight into d_out)
  gemm_bf16nt_kernel<<<dim3((Bc * NGc * Tc) / 128, Cc / 128), 256, 0, stream>>>(
      yB, WpT, out, Bc * NGc * Tc, Cc, Cc);
}
